// GraphConvLayer_22677427323637
// MI455X (gfx1250) — compile-verified, matrix operand order checked by reading, not run on hardware
//
#include <hip/hip_runtime.h>

typedef __attribute__((ext_vector_type(16))) _Float16 v16h;
typedef __attribute__((ext_vector_type(8)))  _Float16 v8h;
typedef __attribute__((ext_vector_type(2)))  _Float16 v2h;
typedef __attribute__((ext_vector_type(8)))  float    v8f;

#define B_TOT   65536
#define NNEI    35
#define C_DIM   64
#define H_DIM   16
#define NROWS   36      // 35 neighbors + 1 target row
#define NPAD    72      // f16 row stride: 144B = 16B-aligned, rotates banks
#define BLOCKS  256
#define WPB     8       // waves per block (wave32)
#define LRELU   0.2f

static __device__ __forceinline__ v16h cat8(v8h lo, v8h hi) {
    return __builtin_shufflevector(lo, hi, 0,1,2,3,4,5,6,7,8,9,10,11,12,13,14,15);
}

__global__ __launch_bounds__(256) void gat_fused_kernel(
    const float* __restrict__ xt,  const float* __restrict__ xn,
    const float* __restrict__ aw,  const float* __restrict__ ab,
    const float* __restrict__ a1w, const float* __restrict__ a1b,
    const float* __restrict__ a2w, const float* __restrict__ a2b,
    float* __restrict__ out)
{
    __shared__ _Float16 sW[C_DIM * C_DIM];        // align_w, f16
    __shared__ _Float16 sA1[H_DIM * 2 * C_DIM];   // attn1_w, f16
    __shared__ float    sAB[C_DIM];
    __shared__ float    sA1B[H_DIM];
    __shared__ float    sA2W[H_DIM];
    __shared__ float    sA2B[1];
    __shared__ _Float16 sN[WPB][NROWS * NPAD];    // per-wave aligned rows (f16)
    __shared__ float    sAtt[WPB][40];            // per-wave attention weights

    const int tid  = threadIdx.x;
    const int w    = tid >> 5;
    const int lane = tid & 31;
    const int lh   = lane & 15;   // lane % 16
    const int hi   = lane >> 4;   // lane half (0/1)
    const int h8   = hi * 8;

    // ---- stage weights into LDS (f32 -> f16) ----
    for (int i = tid; i < C_DIM * C_DIM; i += blockDim.x)     sW[i]  = (_Float16)aw[i];
    for (int i = tid; i < H_DIM * 2 * C_DIM; i += blockDim.x) sA1[i] = (_Float16)a1w[i];
    if (tid < C_DIM)                   sAB[tid]      = ab[tid];
    if (tid < H_DIM)                   sA1B[tid]     = a1b[tid];
    if (tid >= 64 && tid < 64 + H_DIM) sA2W[tid-64]  = a2w[tid-64];
    if (tid == 96)                     sA2B[0]       = a2b[0];
    __syncthreads();

    // ---- persistent A-side fragments (M = channel / hidden unit) ----
    // 16-bit operand layout (ISA 7.12.2): element e of lane L holds
    //   K = e + (e<8?0:8) + (L<16?0:8),  M/N index = L%16
    v16h WA[4][2];                    // align_w rows: D = W x X^T
    for (int dt = 0; dt < 4; ++dt)
        for (int ks = 0; ks < 2; ++ks) {
            int d = dt * 16 + lh;
            int base = 32 * ks + h8;
            v8h lo  = *(const v8h*)&sW[d * C_DIM + base];
            v8h hi8 = *(const v8h*)&sW[d * C_DIM + base + 16];
            WA[dt][ks] = cat8(lo, hi8);
        }
    v16h A1[4];                       // attn1_w rows: D = A1 x a_input^T
    for (int ks = 0; ks < 4; ++ks) {
        int base = 32 * ks + h8;
        v8h lo  = *(const v8h*)&sA1[lh * (2 * C_DIM) + base];
        v8h hi8 = *(const v8h*)&sA1[lh * (2 * C_DIM) + base + 16];
        A1[ks] = cat8(lo, hi8);
    }
    // ---- per-lane constants hoisted out of the b loop ----
    float abias[4][8];
    #pragma unroll
    for (int dt = 0; dt < 4; ++dt)
        #pragma unroll
        for (int r = 0; r < 8; ++r) abias[dt][r] = sAB[dt * 16 + h8 + r];
    float a1bias[8], coef[8];
    #pragma unroll
    for (int r = 0; r < 8; ++r) { a1bias[r] = sA1B[h8 + r]; coef[r] = sA2W[h8 + r]; }
    const float a2bv = sA2B[0];
    _Float16* const myN = &sN[w][0];
    float*    const myA = &sAtt[w][0];

    const int waveGlobal = blockIdx.x * WPB + w;
    const int WAVES = BLOCKS * WPB;               // 2048
    const int ITERS = B_TOT / WAVES;              // 32 (exact)

    for (int it = 0; it < ITERS; ++it) {
        const int b = it * WAVES + waveGlobal;

        // prefetch next iteration's 8960B neighbor block (global_prefetch_b8)
        if (it + 1 < ITERS) {
            const float* nxt = xn + (size_t)(b + WAVES) * NNEI * C_DIM;
            __builtin_prefetch(nxt + lane * 70, 0, 0);
            __builtin_prefetch(nxt + lane * 70 + 32, 0, 0);
        }

        // ========== Phase 1: sN = W x [neigh;target]^T + bias ==========
        v16h Bf[3][2];                 // X^T fragments: lane = neighbor, K = channel
        #pragma unroll
        for (int nt = 0; nt < 3; ++nt) {
            int n = nt * 16 + lh;
            const float* src = (n < NNEI)  ? (xn + ((size_t)b * NNEI + n) * C_DIM)
                             : (n == NNEI) ? (xt + (size_t)b * C_DIM)
                             : nullptr;
            #pragma unroll
            for (int ks = 0; ks < 2; ++ks) {
                int base = 32 * ks + h8;
                v16h f = {};
                if (src) {
                    float4 g0 = *(const float4*)(src + base);
                    float4 g1 = *(const float4*)(src + base + 4);
                    float4 g2 = *(const float4*)(src + base + 16);
                    float4 g3 = *(const float4*)(src + base + 20);
                    f[0]=(_Float16)g0.x;  f[1]=(_Float16)g0.y;  f[2]=(_Float16)g0.z;  f[3]=(_Float16)g0.w;
                    f[4]=(_Float16)g1.x;  f[5]=(_Float16)g1.y;  f[6]=(_Float16)g1.z;  f[7]=(_Float16)g1.w;
                    f[8]=(_Float16)g2.x;  f[9]=(_Float16)g2.y;  f[10]=(_Float16)g2.z; f[11]=(_Float16)g2.w;
                    f[12]=(_Float16)g3.x; f[13]=(_Float16)g3.y; f[14]=(_Float16)g3.z; f[15]=(_Float16)g3.w;
                }
                Bf[nt][ks] = f;
            }
        }
        #pragma unroll
        for (int nt = 0; nt < 3; ++nt) {
            int n = nt * 16 + lh;
            #pragma unroll
            for (int dt = 0; dt < 4; ++dt) {
                v8f acc = {};
                acc = __builtin_amdgcn_wmma_f32_16x16x32_f16(false, WA[dt][0], false, Bf[nt][0], (short)0, acc, false, false);
                acc = __builtin_amdgcn_wmma_f32_16x16x32_f16(false, WA[dt][1], false, Bf[nt][1], (short)0, acc, false, false);
                v8h o;
                #pragma unroll
                for (int r = 0; r < 8; ++r) o[r] = (_Float16)(acc[r] + abias[dt][r]);
                if (n < NROWS)                              // one b128 store per tile
                    *(v8h*)&myN[n * NPAD + dt * 16 + h8] = o;
            }
        }
        __syncthreads();

        // ========== Phase 2: e[n] = leaky(w2 . relu(A1 x a_input^T + b1) + b2) ==========
        v16h T0, T1;                    // target half of a_input: lane-uniform broadcast
        {
            const _Float16* t = &myN[NNEI * NPAD];
            T0 = cat8(*(const v8h*)&t[h8],      *(const v8h*)&t[h8 + 16]);
            T1 = cat8(*(const v8h*)&t[h8 + 32], *(const v8h*)&t[h8 + 48]);
        }
        float ev[3];
        #pragma unroll
        for (int nt = 0; nt < 3; ++nt) {
            int na = nt * 16 + lh;
            int nc = (na < NROWS) ? na : 0;     // clamp (masked below)
            const _Float16* nr = &myN[nc * NPAD];
            v16h N0 = cat8(*(const v8h*)&nr[h8],      *(const v8h*)&nr[h8 + 16]);
            v16h N1 = cat8(*(const v8h*)&nr[h8 + 32], *(const v8h*)&nr[h8 + 48]);
            v8f acc = {};
            acc = __builtin_amdgcn_wmma_f32_16x16x32_f16(false, A1[0], false, T0, (short)0, acc, false, false);
            acc = __builtin_amdgcn_wmma_f32_16x16x32_f16(false, A1[1], false, T1, (short)0, acc, false, false);
            acc = __builtin_amdgcn_wmma_f32_16x16x32_f16(false, A1[2], false, N0, (short)0, acc, false, false);
            acc = __builtin_amdgcn_wmma_f32_16x16x32_f16(false, A1[3], false, N1, (short)0, acc, false, false);
            float s = 0.f;
            #pragma unroll
            for (int r = 0; r < 8; ++r) {
                float hv = acc[r] + a1bias[r];
                hv = hv > 0.f ? hv : 0.f;       // relu
                s += hv * coef[r];              // dot with attn2_w (own 8 hidden units)
            }
            s += __shfl_xor(s, 16);             // other 8 hidden units live cross-half
            float e2 = s + a2bv;
            e2 = e2 > 0.f ? e2 : LRELU * e2;    // leaky relu
            ev[nt] = (na < NNEI) ? e2 : -INFINITY;
        }
        // softmax over 35 scores: 3 per lane across lanes 0..15 (halves replicated)
        float mx = fmaxf(ev[0], fmaxf(ev[1], ev[2]));
        mx = fmaxf(mx, __shfl_xor(mx, 1));
        mx = fmaxf(mx, __shfl_xor(mx, 2));
        mx = fmaxf(mx, __shfl_xor(mx, 4));
        mx = fmaxf(mx, __shfl_xor(mx, 8));
        float p[3];
        float sum = 0.f;
        #pragma unroll
        for (int nt = 0; nt < 3; ++nt) { p[nt] = __expf(ev[nt] - mx); sum += p[nt]; }
        sum += __shfl_xor(sum, 1);
        sum += __shfl_xor(sum, 2);
        sum += __shfl_xor(sum, 4);
        sum += __shfl_xor(sum, 8);
        float inv = 1.0f / sum;
        if (hi == 0) {
            #pragma unroll
            for (int nt = 0; nt < 3; ++nt) {
                int na = nt * 16 + lh;
                if (na < NNEI) myA[na] = p[nt] * inv;
            }
        }
        __syncthreads();

        // ========== Phase 3: h' = sum_n a[n]*n_aligned[n,:], ELU ==========
        int c0 = lane * 2;
        float acc0 = 0.f, acc1 = 0.f;
        #pragma unroll 5
        for (int n = 0; n < NNEI; ++n) {
            float a = myA[n];
            v2h t2 = *(const v2h*)&myN[n * NPAD + c0];   // one b32 load
            acc0 += a * (float)t2[0];
            acc1 += a * (float)t2[1];
        }
        acc0 = acc0 > 0.f ? acc0 : (__expf(acc0) - 1.0f);   // elu
        acc1 = acc1 > 0.f ? acc1 : (__expf(acc1) - 1.0f);
        *(float2*)(out + (size_t)b * C_DIM + c0) = make_float2(acc0, acc1);
        __syncthreads();   // protect per-wave LDS tiles before next iteration
    }
}

extern "C" void kernel_launch(void* const* d_in, const int* in_sizes, int n_in,
                              void* d_out, int out_size, void* d_ws, size_t ws_size,
                              hipStream_t stream) {
    const float* xt  = (const float*)d_in[0];
    const float* xn  = (const float*)d_in[1];
    const float* aw  = (const float*)d_in[2];
    const float* ab  = (const float*)d_in[3];
    const float* a1w = (const float*)d_in[4];
    const float* a1b = (const float*)d_in[5];
    const float* a2w = (const float*)d_in[6];
    const float* a2b = (const float*)d_in[7];
    float* o = (float*)d_out;
    gat_fused_kernel<<<BLOCKS, WPB * 32, 0, stream>>>(xt, xn, aw, ab, a1w, a1b, a2w, a2b, o);
}
